// ChordMixerNet_46660524704301
// MI455X (gfx1250) — matrix operand order repaired and checked by reading, not compiled, not run on hardware
//
#include <hip/hip_runtime.h>
#include <hip/hip_bf16.h>
#include <math.h>

// ---------------- problem constants ----------------
#define S   131072
#define T   17
#define D   272
#define H   544
#define L   17
#define NC  10

// ---------------- tiling constants -----------------
#define MTILE     64             // rows per block
#define BLOCK     128            // 4 waves (wave32)
#define KA        288            // D padded to multiple of 32
#define A_PITCH   296            // bf16 elems per A row (16B-aligned rows)
#define HID_PITCH 552            // bf16 elems per hidden row
#define B_PITCH1  296            // packed W1 tile row pitch (k extent 288)
#define B_PITCH2  552            // packed W2 tile row pitch (k extent 544)

#define NT1 (H / 16)             // 34 n-tiles in GEMM1
#define NT2 T                    // 17 n-tiles (== tracks) in GEMM2

#define W1_TILE_ELEMS (16 * B_PITCH1)          // 4736 bf16 = 9472 B
#define W2_TILE_ELEMS (16 * B_PITCH2)          // 8832 bf16 = 17664 B
#define W1_TILE_QW    ((W1_TILE_ELEMS * 2) / 8) // 1184
#define W2_TILE_QW    ((W2_TILE_ELEMS * 2) / 8) // 2208

#define LDS_A_OFF   0
#define LDS_A_BYTES (MTILE * A_PITCH * 2)      // 37888
#define LDS_H_OFF   (LDS_A_OFF + LDS_A_BYTES)
#define LDS_H_BYTES (MTILE * HID_PITCH * 2)    // 70656
#define LDS_B0_OFF  (LDS_H_OFF + LDS_H_BYTES)  // 108544
#define LDS_B_BYTES (16 * B_PITCH2 * 2)        // 17664
#define LDS_B1_OFF  (LDS_B0_OFF + LDS_B_BYTES) // 126208
#define SMEM_BYTES  (LDS_B1_OFF + LDS_B_BYTES) // 143872  (< 160KB CU-mode LDS)

typedef __attribute__((ext_vector_type(16))) __bf16    bf16x16;
typedef __attribute__((ext_vector_type(8)))  __bf16    bf16x8;
typedef __attribute__((ext_vector_type(4)))  __bf16    bf16x4;
typedef __attribute__((ext_vector_type(8)))  float     f32x8;
typedef __attribute__((ext_vector_type(4)))  float     f32x4;
typedef __attribute__((ext_vector_type(4)))  unsigned  u32x4;
typedef __attribute__((ext_vector_type(8)))  int       i32x8;
typedef __attribute__((ext_vector_type(4)))  int       i32x4;

#if defined(__gfx1250__) && __has_builtin(__builtin_amdgcn_tensor_load_to_lds) && \
    __has_builtin(__builtin_amdgcn_s_wait_tensorcnt)
#define USE_TDM 1
#else
#define USE_TDM 0
#endif

__device__ __forceinline__ unsigned short f2bfu(float f) {
    unsigned u = __builtin_bit_cast(unsigned, f);
    unsigned r = (u + 0x7FFFu + ((u >> 16) & 1u)) >> 16;   // RNE
    return (unsigned short)r;
}
__device__ __forceinline__ __bf16 f2bf(float f) {
    unsigned short s = f2bfu(f);
    return __builtin_bit_cast(__bf16, s);
}

// gather 16 bf16 as two 16B LDS loads (both pointers 16B aligned)
__device__ __forceinline__ bf16x16 ld16(const __bf16* p0, const __bf16* p1) {
    bf16x8 lo = *(const bf16x8*)p0;
    bf16x8 hi = *(const bf16x8*)p1;
    return __builtin_shufflevector(lo, hi, 0,1,2,3,4,5,6,7,8,9,10,11,12,13,14,15);
}

__device__ __forceinline__ float gelu_exact(float x) {
    return 0.5f * x * (1.0f + erff(x * 0.70710678118654752f));
}

#if USE_TDM
// 1-D contiguous TDM copy: `qw` qwords from global `gsrc` into LDS byte offset
// `lds_off`. Builds the D# (group0 128b + group1 256b, groups 2/3 zero) and
// issues TENSOR_LOAD_TO_LDS; completion tracked with TENSORcnt.
__device__ __forceinline__ void tdm_copy(unsigned lds_off, const void* gsrc, unsigned qw) {
    unsigned long long ga = (unsigned long long)(size_t)gsrc;
    u32x4 g0;
    g0[0] = 1u;                                            // count=1, user descriptor
    g0[1] = lds_off;                                       // lds_addr (bytes)
    g0[2] = (unsigned)(ga & 0xffffffffu);                  // global_addr[31:0]
    g0[3] = (unsigned)((ga >> 32) & 0x01ffffffu) | (2u << 30); // addr[56:32] | type=2
    i32x8 g1;
    g1[0] = (int)(3u << 16);                               // wg_mask=0, data_size=8B
    g1[1] = (int)((qw & 0xffffu) << 16);                   // tensor_dim0[15:0]
    g1[2] = (int)(((qw >> 16) & 0xffffu) | (1u << 16));    // tensor_dim0[31:16], tensor_dim1=1
    g1[3] = (int)((qw & 0xffffu) << 16);                   // tile_dim0 = qw
    g1[4] = 0;                                             // tile_dim1/2 unused
    g1[5] = (int)qw;                                       // tensor_dim0_stride[31:0]
    g1[6] = 0;
    g1[7] = 0;
    i32x4 z4 = {0, 0, 0, 0};
#if defined(__clang_major__) && __clang_major__ >= 23
    i32x8 z8 = {0, 0, 0, 0, 0, 0, 0, 0};
    __builtin_amdgcn_tensor_load_to_lds(g0, g1, z4, z4, z8, 0);
#else
    __builtin_amdgcn_tensor_load_to_lds(g0, g1, z4, z4, 0);
#endif
}
#endif

// ---------------------------------------------------------------------------
// Fused layer kernel. Weights arrive pre-packed bf16 in LDS tile layout:
//   w1t: [NT1][16][B_PITCH1]  (k padded 272->288 with zeros)
//   w2t: [NT2][16][B_PITCH2]
// Per-track cyclic roll folded into GEMM2 epilogue as scatter + residual.
// ---------------------------------------------------------------------------
__global__ void __launch_bounds__(BLOCK)
chordmixer_layer_k(const float* __restrict__ xin, float* __restrict__ xout,
                   const unsigned short* __restrict__ w1tu,
                   const float* __restrict__ b1,
                   const unsigned short* __restrict__ w2tu,
                   const float* __restrict__ b2)
{
    extern __shared__ char smem[];
    __bf16* lds_a   = (__bf16*)(smem + LDS_A_OFF);
    __bf16* lds_hid = (__bf16*)(smem + LDS_H_OFF);
    const __bf16* w1t = (const __bf16*)w1tu;
    const __bf16* w2t = (const __bf16*)w2tu;

    const int tid  = threadIdx.x;
    const int w    = tid >> 5;          // wave id 0..3
    const int lane = tid & 31;
    const int g    = lane >> 4;         // half-wave
    const int m16  = lane & 15;         // M index for A / N index for B,C,D
    const int row0 = blockIdx.x * MTILE;

#if USE_TDM
    if (w == 0) tdm_copy(LDS_B0_OFF, w1t, W1_TILE_QW);   // prefetch W1 tile 0
#endif

    // ---- stage x rows -> bf16 LDS (vectorized; K padded 272->288) ----
    for (int i = tid; i < MTILE * (KA / 4); i += BLOCK) {
        int r = i / (KA / 4), c4 = (i % (KA / 4)) * 4;
        f32x4 v = {};
        if (c4 < D) v = *(const f32x4*)&xin[(size_t)(row0 + r) * D + c4];
        bf16x4 o;
        o[0] = f2bf(v[0]); o[1] = f2bf(v[1]); o[2] = f2bf(v[2]); o[3] = f2bf(v[3]);
        *(bf16x4*)&lds_a[r * A_PITCH + c4] = o;
    }
    __syncthreads();

    // ---- preload A fragments (16x32 bf16, 9 k-chunks) ----
    bf16x16 afrag[9];
    {
        const __bf16* arow = lds_a + (w * 16 + m16) * A_PITCH;
        #pragma unroll
        for (int kc = 0; kc < 9; ++kc) {
            const __bf16* p = arow + kc * 32 + g * 8;
            afrag[kc] = ld16(p, p + 16);
        }
    }

    // ================= GEMM1: hid = GELU(x @ W1 + b1) =================
    for (int nt = 0; nt < NT1; ++nt) {
        const int cur = nt & 1;
#if USE_TDM
        if (w == 0) __builtin_amdgcn_s_wait_tensorcnt(0);  // tile nt landed
        __syncthreads();                                   // publish; old buf free
        if (w == 0 && nt + 1 < NT1)
            tdm_copy(cur ? LDS_B0_OFF : LDS_B1_OFF,
                     w1t + (size_t)(nt + 1) * W1_TILE_ELEMS, W1_TILE_QW);
        const __bf16* bt = (const __bf16*)(smem + (cur ? LDS_B1_OFF : LDS_B0_OFF));
#else
        __syncthreads();
        {
            const __bf16* src = w1t + (size_t)nt * W1_TILE_ELEMS;
            __bf16* dst = (__bf16*)(smem + LDS_B0_OFF);
            for (int i = tid * 8; i < W1_TILE_ELEMS; i += BLOCK * 8)
                *(bf16x8*)(dst + i) = *(const bf16x8*)(src + i);
        }
        __syncthreads();
        const __bf16* bt = (const __bf16*)(smem + LDS_B0_OFF);
#endif
        f32x8 acc = {};
        #pragma unroll
        for (int kc = 0; kc < 9; ++kc) {
            const __bf16* bb = bt + m16 * B_PITCH1 + kc * 32 + g * 8;
            bf16x16 bfrag = ld16(bb, bb + 16);
            acc = __builtin_amdgcn_wmma_f32_16x16x32_bf16(
                      false, afrag[kc], false, bfrag, (short)0, acc, false, false);
        }
        float b1v = b1[nt * 16 + m16];
        #pragma unroll
        for (int r = 0; r < 8; ++r) {
            float h = gelu_exact(acc[r] + b1v);
            lds_hid[(w * 16 + r + 8 * g) * HID_PITCH + nt * 16 + m16] = f2bf(h);
        }
    }

    // ============ GEMM2 + per-track roll + residual scatter ============
    __syncthreads();                         // everyone done with GEMM1 buffers
#if USE_TDM
    if (w == 0) tdm_copy(LDS_B0_OFF, w2t, W2_TILE_QW);     // prefetch W2 tile 0
#endif
    const __bf16* hrow = lds_hid + (w * 16 + m16) * HID_PITCH;
    for (int nt2 = 0; nt2 < NT2; ++nt2) {    // n-tile == track
        const int cur = nt2 & 1;
#if USE_TDM
        if (w == 0) __builtin_amdgcn_s_wait_tensorcnt(0);
        __syncthreads();
        if (w == 0 && nt2 + 1 < NT2)
            tdm_copy(cur ? LDS_B0_OFF : LDS_B1_OFF,
                     w2t + (size_t)(nt2 + 1) * W2_TILE_ELEMS, W2_TILE_QW);
        const __bf16* bt = (const __bf16*)(smem + (cur ? LDS_B1_OFF : LDS_B0_OFF));
#else
        __syncthreads();
        {
            const __bf16* src = w2t + (size_t)nt2 * W2_TILE_ELEMS;
            __bf16* dst = (__bf16*)(smem + LDS_B0_OFF);
            for (int i = tid * 8; i < W2_TILE_ELEMS; i += BLOCK * 8)
                *(bf16x8*)(dst + i) = *(const bf16x8*)(src + i);
        }
        __syncthreads();
        const __bf16* bt = (const __bf16*)(smem + LDS_B0_OFF);
#endif
        f32x8 acc = {};
        #pragma unroll
        for (int kc = 0; kc < H / 32; ++kc) {
            const __bf16* pa = hrow + kc * 32 + g * 8;
            bf16x16 af = ld16(pa, pa + 16);
            const __bf16* bb = bt + m16 * B_PITCH2 + kc * 32 + g * 8;
            bf16x16 bf_ = ld16(bb, bb + 16);
            acc = __builtin_amdgcn_wmma_f32_16x16x32_bf16(
                      false, af, false, bf_, (short)0, acc, false, false);
        }
        float b2v   = b2[nt2 * 16 + m16];
        int   shift = nt2 ? (1 << (nt2 - 1)) : 0;
        #pragma unroll
        for (int r = 0; r < 8; ++r) {
            int gr = row0 + w * 16 + r + 8 * g;
            int rp = (int)(((unsigned)(gr - shift)) & (unsigned)(S - 1));
            size_t o = (size_t)rp * D + nt2 * 16 + m16;
            xout[o] = xin[o] + acc[r] + b2v;   // residual + rolled track chunk
        }
    }
}

// ---------------------------------------------------------------------------
// One-time weight pre-pack: f32 -> bf16, LDS tile layout, K zero-padded.
// ---------------------------------------------------------------------------
__global__ void pack_w1_k(const float* __restrict__ W1s, unsigned short* __restrict__ w1p)
{
    unsigned i = blockIdx.x * 256u + threadIdx.x;
    if (i >= (unsigned)L * NT1 * W1_TILE_ELEMS) return;
    unsigned k  = i % B_PITCH1;
    unsigned n  = (i / B_PITCH1) & 15u;
    unsigned nt = (i / (B_PITCH1 * 16)) % NT1;
    unsigned l  = i / (B_PITCH1 * 16 * NT1);
    float v = (k < D) ? W1s[((size_t)l * D + k) * H + nt * 16 + n] : 0.0f;
    w1p[i] = f2bfu(v);
}

__global__ void pack_w2_k(const float* __restrict__ W2s, unsigned short* __restrict__ w2p)
{
    unsigned i = blockIdx.x * 256u + threadIdx.x;
    if (i >= (unsigned)L * NT2 * W2_TILE_ELEMS) return;
    unsigned k  = i % B_PITCH2;
    unsigned n  = (i / B_PITCH2) & 15u;
    unsigned nt = (i / (B_PITCH2 * 16)) % NT2;
    unsigned l  = i / (B_PITCH2 * 16 * NT2);
    float v = (k < H) ? W2s[((size_t)l * H + k) * D + nt * 16 + n] : 0.0f;
    w2p[i] = f2bfu(v);
}

// ---------------------------------------------------------------------------
// Embedding:  x0[s, :] = emb[data[s], :]   (float4 vectorized, D % 4 == 0)
// ---------------------------------------------------------------------------
__global__ void embed_k(const int* __restrict__ data, const float* __restrict__ emb,
                        float* __restrict__ x0)
{
    unsigned i = blockIdx.x * 256u + threadIdx.x;   // over S * (D/4)
    if (i >= (unsigned)S * (D / 4)) return;
    unsigned s = i / (D / 4), c4 = (i % (D / 4)) * 4u;
    f32x4 v = *(const f32x4*)&emb[(size_t)data[s] * D + c4];
    *(f32x4*)&x0[(size_t)s * D + c4] = v;
}

// ---------------------------------------------------------------------------
// Stage-1 reduction: block b sums rows [128b, 128b+128) per column
// ---------------------------------------------------------------------------
__global__ void rowsum_k(const float* __restrict__ x, float* __restrict__ part)
{
    int c = threadIdx.x;
    if (c >= D) return;
    float a = 0.0f;
    size_t base = (size_t)blockIdx.x * 128 * D;
    for (int r = 0; r < 128; ++r) a += x[base + (size_t)r * D + c];
    part[(size_t)blockIdx.x * D + c] = a;
}

// ---------------------------------------------------------------------------
// Stage-2: mean over S, then 272x10 projection
// ---------------------------------------------------------------------------
__global__ void final_k(const float* __restrict__ part, const float* __restrict__ Wf,
                        const float* __restrict__ bf_, float* __restrict__ out)
{
    __shared__ float mean[D];
    int t = threadIdx.x;
    if (t < D) {
        float a = 0.0f;
        for (int i = 0; i < S / 128; ++i) a += part[(size_t)i * D + t];
        mean[t] = a * (1.0f / (float)S);
    }
    __syncthreads();
    if (t < NC) {
        float a = bf_[t];
        for (int d = 0; d < D; ++d) a += mean[d] * Wf[d * NC + t];
        out[t] = a;
    }
}

// ---------------------------------------------------------------------------
extern "C" void kernel_launch(void* const* d_in, const int* in_sizes, int n_in,
                              void* d_out, int out_size, void* d_ws, size_t ws_size,
                              hipStream_t stream)
{
    (void)in_sizes; (void)n_in; (void)out_size; (void)ws_size;

    const int*   data = (const int*)  d_in[0];
    const float* emb  = (const float*)d_in[1];
    const float* W1s  = (const float*)d_in[2];
    const float* b1s  = (const float*)d_in[3];
    const float* W2s  = (const float*)d_in[4];
    const float* b2s  = (const float*)d_in[5];
    const float* Wf   = (const float*)d_in[6];
    const float* bfv  = (const float*)d_in[7];
    float*       out  = (float*)d_out;

    // workspace carve: two x buffers + partials + packed bf16 weights
    float* buf0 = (float*)d_ws;
    float* buf1 = buf0 + (size_t)S * D;
    float* part = buf1 + (size_t)S * D;
    unsigned short* w1p = (unsigned short*)(part + (size_t)(S / 128) * D);
    unsigned short* w2p = w1p + (size_t)L * NT1 * W1_TILE_ELEMS;

    (void)hipFuncSetAttribute((const void*)chordmixer_layer_k,
                              hipFuncAttributeMaxDynamicSharedMemorySize, SMEM_BYTES);

    // one-time weight pre-pack (bf16, LDS tile layout)
    {
        unsigned n1 = (unsigned)L * NT1 * W1_TILE_ELEMS;
        unsigned n2 = (unsigned)L * NT2 * W2_TILE_ELEMS;
        pack_w1_k<<<(n1 + 255) / 256, 256, 0, stream>>>(W1s, w1p);
        pack_w2_k<<<(n2 + 255) / 256, 256, 0, stream>>>(W2s, w2p);
    }

    // x0 = emb[data]
    {
        unsigned n = (unsigned)S * (D / 4);
        embed_k<<<(n + 255) / 256, 256, 0, stream>>>(data, emb, buf0);
    }

    // 17 fused MLP + roll + residual layers, double-buffered
    float* cur = buf0;
    float* nxt = buf1;
    for (int l = 0; l < L; ++l) {
        chordmixer_layer_k<<<S / MTILE, BLOCK, SMEM_BYTES, stream>>>(
            cur, nxt,
            w1p + (size_t)l * NT1 * W1_TILE_ELEMS, b1s + (size_t)l * H,
            w2p + (size_t)l * NT2 * W2_TILE_ELEMS, b2s + (size_t)l * D);
        float* t = cur; cur = nxt; nxt = t;
    }

    // mean over S  ->  [D]  ->  @Wf + bf  ->  [10]
    rowsum_k<<<S / 128, 288, 0, stream>>>(cur, part);
    final_k<<<1, 288, 0, stream>>>(part, Wf, bfv, out);
}
